// PyramidLevel1Block_76656576299155
// MI455X (gfx1250) — matrix operand order repaired and checked by reading, not compiled, not run on hardware
//
#include <hip/hip_runtime.h>

// ---------------------------------------------------------------------------
// Problem constants (from reference setup_inputs)
// ---------------------------------------------------------------------------
#define BATCH   4
#define M_DIM   65536
#define N_DIM   16384
#define C_PRE   64
#define C_CUR   128
#define C_IN    192
#define D_OUT   128
#define BN_EPS  1e-5f

#define M_TILE  64          // M rows per block
#define KP      200         // padded LDS row pitch in halfwords (400 B)

typedef __attribute__((ext_vector_type(16))) __bf16 bf16x16;
typedef __attribute__((ext_vector_type(8)))  __bf16 bf16x8;
typedef __attribute__((ext_vector_type(8)))  float  f32x8;

// round-to-nearest-even f32 -> bf16 bits
static __device__ __forceinline__ unsigned short bf16_rn(float f) {
    unsigned int u = __float_as_uint(f);
    u += 0x7FFFu + ((u >> 16) & 1u);
    return (unsigned short)(u >> 16);
}
static __device__ __forceinline__ void split_bf16(float f, unsigned short& hi,
                                                  unsigned short& lo) {
    hi = bf16_rn(f);
    float fh = __uint_as_float(((unsigned int)hi) << 16);
    lo = bf16_rn(f - fh);
}
static __device__ __forceinline__ bf16x16 cat8(bf16x8 x, bf16x8 y) {
    bf16x16 r;
#pragma unroll
    for (int i = 0; i < 8; ++i) { r[i] = x[i]; r[i + 8] = y[i]; }
    return r;
}

// CDNA5 async global->LDS copy (16 B per lane), tracked by ASYNCcnt.
// dstLds: wave-relative LDS byte address (low 32 bits of generic LDS pointer).
static __device__ __forceinline__ void async_to_lds_b128(const void* src,
                                                         void* dstLds) {
    unsigned int dst = (unsigned int)(uintptr_t)dstLds;
    unsigned long long gaddr = (unsigned long long)(uintptr_t)src;
    asm volatile("global_load_async_to_lds_b128 %0, %1, off"
                 :: "v"(dst), "v"(gaddr) : "memory");
}
static __device__ __forceinline__ void wait_asynccnt0() {
    asm volatile("s_wait_asynccnt 0" ::: "memory");
}

// ---------------------------------------------------------------------------
// Pass 1a: transpose cur_x (B, C_CUR, N) -> feat planes (B, N, C_CUR) as
// bf16 hi/lo. Makes each gathered row a contiguous 256 B read.
// ---------------------------------------------------------------------------
__global__ __launch_bounds__(256) void transpose_feat_kernel(
    const float* __restrict__ cur_x,
    unsigned short* __restrict__ featHi,
    unsigned short* __restrict__ featLo) {
    __shared__ float tile[32][33];
    const int b  = blockIdx.z;
    const int nB = blockIdx.x * 32;
    const int cB = blockIdx.y * 32;
    const int tx = threadIdx.x, ty = threadIdx.y;   // 32 x 8

#pragma unroll
    for (int i = 0; i < 4; ++i) {
        int c = cB + ty + i * 8;
        tile[ty + i * 8][tx] =
            cur_x[((size_t)(b * C_CUR + c)) * N_DIM + nB + tx];
    }
    __syncthreads();
#pragma unroll
    for (int i = 0; i < 4; ++i) {
        int n   = nB + ty + i * 8;
        float f = tile[tx][ty + i * 8];
        unsigned short h, l;
        split_bf16(f, h, l);
        size_t o = ((size_t)b * N_DIM + n) * C_CUR + cB + tx;
        featHi[o] = h;
        featLo[o] = l;
    }
}

// ---------------------------------------------------------------------------
// Pass 1b: fold BN into weights/bias; transpose w to [d][k] bf16 hi/lo.
// ---------------------------------------------------------------------------
__global__ __launch_bounds__(256) void prep_weights_kernel(
    const float* __restrict__ w,     const float* __restrict__ bias,
    const float* __restrict__ gamma, const float* __restrict__ beta,
    const float* __restrict__ mean,  const float* __restrict__ var,
    unsigned short* __restrict__ wHiG, unsigned short* __restrict__ wLoG,
    float* __restrict__ biasG) {
    __shared__ float sS[D_OUT];
    const int tid = threadIdx.x;
    if (tid < D_OUT) {
        float s  = __frsqrt_rn(var[tid] + BN_EPS) * gamma[tid];
        sS[tid]  = s;
        biasG[tid] = (bias[tid] - mean[tid]) * s + beta[tid];
    }
    __syncthreads();
    for (int i = tid; i < C_IN * D_OUT; i += 256) {
        int c = i / D_OUT, d = i % D_OUT;
        float f = w[i] * sS[d];
        unsigned short h, l;
        split_bf16(f, h, l);
        wHiG[d * C_IN + c] = h;
        wLoG[d * C_IN + c] = l;
    }
}

// ---------------------------------------------------------------------------
// Pass 2: fused gather + split-bf16 WMMA GEMM + bias + ReLU.
// Block = 256 threads (8 waves), tile = 64(M) x 128(D), K = 192.
// Weights and gathered feat rows stream memory->LDS via async copies.
// ---------------------------------------------------------------------------
__global__ __launch_bounds__(256) void fused_gemm_kernel(
    const float* __restrict__ pre_x,
    const long long* __restrict__ up_idx,
    const unsigned short* __restrict__ featHi,
    const unsigned short* __restrict__ featLo,
    const unsigned short* __restrict__ wHiG,
    const unsigned short* __restrict__ wLoG,
    const float* __restrict__ biasG,
    float* __restrict__ out) {
    extern __shared__ char smem[];
    unsigned short* aHi = (unsigned short*)smem;          // [M_TILE][KP]
    unsigned short* aLo = aHi + M_TILE * KP;
    unsigned short* wHi = aLo + M_TILE * KP;              // [D_OUT][KP]
    unsigned short* wLo = wHi + D_OUT * KP;
    float* biasS = (float*)(wLo + D_OUT * KP);

    const int tid = threadIdx.x;
    const int b   = blockIdx.y;
    const int m0  = blockIdx.x * M_TILE;

    if (tid < D_OUT) biasS[tid] = biasG[tid];

    // async-stage weights (B matrix, [d][k], k contiguous) hi/lo into LDS
    for (int i = tid; i < (D_OUT * C_IN) / 8; i += 256) {
        int flat = i * 8;                        // halfword index, 8 per b128
        int d = flat / C_IN, k = flat % C_IN;
        async_to_lds_b128(wHiG + flat, &wHi[d * KP + k]);
        async_to_lds_b128(wLoG + flat, &wLo[d * KP + k]);
    }

    // async-stage A cols 64..191: gather transposed feat rows (256 B each)
    {
        int r  = tid >> 2;   // 0..63 : tile row
        int t4 = tid & 3;    // 4 threads per row
        long long n = up_idx[(size_t)b * M_DIM + m0 + r];
        size_t rowOff = ((size_t)b * N_DIM + (size_t)n) * C_CUR;
#pragma unroll
        for (int j = 0; j < 4; ++j) {
            int c16 = t4 * 32 + j * 8;                 // halfword offset
            async_to_lds_b128(featHi + rowOff + c16,
                              &aHi[r * KP + C_PRE + c16]);
            async_to_lds_b128(featLo + rowOff + c16,
                              &aLo[r * KP + C_PRE + c16]);
        }
    }

    // stage A cols 0..63 from pre_x (needs f32 -> bf16 hi/lo split in VALU;
    // this work overlaps the outstanding async copies above)
    for (int i = tid; i < M_TILE * C_PRE; i += 256) {
        int m = i & (M_TILE - 1), c = i >> 6;
        float f = pre_x[((size_t)(b * C_PRE + c)) * M_DIM + m0 + m];
        unsigned short h, l;
        split_bf16(f, h, l);
        aHi[m * KP + c] = h;
        aLo[m * KP + c] = l;
    }

    wait_asynccnt0();      // this wave's async LDS writes have landed
    __syncthreads();       // cross-wave visibility

    const int wave    = tid >> 5;
    const int lane    = tid & 31;
    const int col     = lane & 15;    // A row-in-subtile / B column (d)
    const int hiHalf  = lane >> 4;    // lane half selects K sub-group
    const int dBase   = wave * 16;

    f32x8 acc[4] = {};
#pragma unroll
    for (int kk = 0; kk < C_IN / 32; ++kk) {
        const int kb = kk * 32;
        // B fragment: lane col = d, K = kb + hiHalf*8 .. +7 and kb + 16 + hiHalf*8 .. +7
        const bf16x8* bph = (const bf16x8*)&wHi[(dBase + col) * KP + kb + hiHalf * 8];
        const bf16x8* bpl = (const bf16x8*)&wLo[(dBase + col) * KP + kb + hiHalf * 8];
        bf16x16 bh = cat8(bph[0], bph[2]);
        bf16x16 bl = cat8(bpl[0], bpl[2]);
#pragma unroll
        for (int ms = 0; ms < 4; ++ms) {
            const int row = ms * 16 + col;
            const int g0  = kb + hiHalf * 8;        // K group 0 (8 halfwords)
            const int g1  = kb + 16 + hiHalf * 8;   // K group 1
            bf16x16 ah = cat8(*(const bf16x8*)&aHi[row * KP + g0],
                              *(const bf16x8*)&aHi[row * KP + g1]);
            bf16x16 al = cat8(*(const bf16x8*)&aLo[row * KP + g0],
                              *(const bf16x8*)&aLo[row * KP + g1]);
            acc[ms] = __builtin_amdgcn_wmma_f32_16x16x32_bf16(
                false, ah, false, bh, (short)0, acc[ms], false, false);
            acc[ms] = __builtin_amdgcn_wmma_f32_16x16x32_bf16(
                false, ah, false, bl, (short)0, acc[ms], false, false);
            acc[ms] = __builtin_amdgcn_wmma_f32_16x16x32_bf16(
                false, al, false, bh, (short)0, acc[ms], false, false);
        }
    }

    // epilogue: bias + ReLU, write (B, M, 128) f32
    const float bb = biasS[dBase + col];
#pragma unroll
    for (int ms = 0; ms < 4; ++ms) {
#pragma unroll
        for (int r = 0; r < 8; ++r) {
            int m   = m0 + ms * 16 + hiHalf * 8 + r;
            float v = acc[ms][r] + bb;
            v = fmaxf(v, 0.0f);
            out[((size_t)b * M_DIM + m) * D_OUT + dBase + col] = v;
        }
    }
}

// ---------------------------------------------------------------------------
// Launch
// ---------------------------------------------------------------------------
extern "C" void kernel_launch(void* const* d_in, const int* in_sizes, int n_in,
                              void* d_out, int out_size, void* d_ws, size_t ws_size,
                              hipStream_t stream) {
    const float*     pre_x  = (const float*)d_in[0];
    const float*     cur_x  = (const float*)d_in[1];
    const long long* up_idx = (const long long*)d_in[2];
    const float*     w      = (const float*)d_in[3];
    const float*     bias   = (const float*)d_in[4];
    const float*     gamma  = (const float*)d_in[5];
    const float*     beta   = (const float*)d_in[6];
    const float*     mean   = (const float*)d_in[7];
    const float*     var    = (const float*)d_in[8];
    float*           out    = (float*)d_out;

    // workspace layout
    const size_t featElems = (size_t)BATCH * N_DIM * C_CUR;      // 8M halfwords
    unsigned short* featHi = (unsigned short*)d_ws;
    unsigned short* featLo = featHi + featElems;
    unsigned short* wHiG   = featLo + featElems;
    unsigned short* wLoG   = wHiG + (size_t)D_OUT * C_IN;
    float*          biasG  = (float*)(wLoG + (size_t)D_OUT * C_IN);

    // Pass 1a: transpose + bf16 split of cur_x
    transpose_feat_kernel<<<dim3(N_DIM / 32, C_CUR / 32, BATCH),
                            dim3(32, 8, 1), 0, stream>>>(cur_x, featHi, featLo);
    // Pass 1b: BN-folded, transposed, bf16-split weights
    prep_weights_kernel<<<1, 256, 0, stream>>>(w, bias, gamma, beta, mean, var,
                                               wHiG, wLoG, biasG);

    // Pass 2: fused gather + WMMA GEMM + bias + ReLU
    const size_t smemBytes =
        (size_t)(2 * M_TILE * KP + 2 * D_OUT * KP) * sizeof(unsigned short) +
        D_OUT * sizeof(float);   // ~150.5 KB, 2 blocks per 320 KB WGP
    (void)hipFuncSetAttribute((const void*)fused_gemm_kernel,
                              hipFuncAttributeMaxDynamicSharedMemorySize,
                              (int)smemBytes);
    fused_gemm_kernel<<<dim3(M_DIM / M_TILE, BATCH, 1), 256, smemBytes, stream>>>(
        pre_x, up_idx, featHi, featLo, wHiG, wLoG, biasG, out);
}
